// LocalLinear_48713519071672
// MI455X (gfx1250) — matrix operand order repaired and checked by reading, not compiled, not run on hardware
//
#include <hip/hip_runtime.h>

typedef __attribute__((ext_vector_type(2)))  float          v2f;
typedef __attribute__((ext_vector_type(4)))  float          v4f;
typedef __attribute__((ext_vector_type(2)))  __bf16         v2bf;
typedef __attribute__((ext_vector_type(4)))  __bf16         v4bf;
typedef __attribute__((ext_vector_type(8)))  unsigned short v8us;
typedef __attribute__((ext_vector_type(16))) unsigned short v16us;
typedef __attribute__((ext_vector_type(16))) __bf16         v16bf;
typedef __attribute__((ext_vector_type(8)))  float          v8f;
typedef __attribute__((ext_vector_type(2)))  unsigned       uv2;

#define IN_F   4096
#define KSZ    64
#define LFEAT  32
#define FOLDS  4096
#define MTILE  128      // batch rows per block
#define FPB    16       // consecutive folds per block (x window shared)
#define XS     88       // bf16 elems per x row slot (176B: 16B aligned)
#define WSTR   72       // bf16 elems per W column slot (144B: 16B aligned)

// ---- split via hardware bf16 converts (v_cvt_pk_bf16_f32 class) ----
__device__ __forceinline__ void split2(float v0, float v1, unsigned& hp, unsigned& lp) {
    v2f v = {v0, v1};
    v2bf h = __builtin_convertvector(v, v2bf);
    v2f hf = __builtin_convertvector(h, v2f);
    v2bf l = __builtin_convertvector(v - hf, v2bf);
    hp = __builtin_bit_cast(unsigned, h);
    lp = __builtin_bit_cast(unsigned, l);
}
__device__ __forceinline__ void split4(v4f v, uv2& hp, uv2& lp) {
    v4bf h = __builtin_convertvector(v, v4bf);
    v4f hf = __builtin_convertvector(h, v4f);
    v4bf l = __builtin_convertvector(v - hf, v4bf);
    hp = __builtin_bit_cast(uv2, h);
    lp = __builtin_bit_cast(uv2, l);
}

__device__ __forceinline__ v16bf cat16(v8us a, v8us b) {
    v16us u = __builtin_shufflevector(a, b, 0,1,2,3,4,5,6,7,8,9,10,11,12,13,14,15);
    return __builtin_bit_cast(v16bf, u);
}
__device__ __forceinline__ v16bf ldB(const unsigned short* p) {
    return cat16(*(const v8us*)p, *(const v8us*)(p + 16));
}
// A fragment from lane-resident row chunks; S = d + kk (compile-time constant)
template<int S>
__device__ __forceinline__ v16bf afrag(const v8us X[9]) {
    constexpr int Q = S >> 3, R = S & 7;
    v8us lo = __builtin_shufflevector(X[Q],     X[Q + 1], R,R+1,R+2,R+3,R+4,R+5,R+6,R+7);
    v8us hi = __builtin_shufflevector(X[Q + 2], X[Q + 3], R,R+1,R+2,R+3,R+4,R+5,R+6,R+7);
    return cat16(lo, hi);
}
__device__ __forceinline__ v8f bwmma(v16bf a, v16bf b, v8f c) {
    return __builtin_amdgcn_wmma_f32_16x16x32_bf16(false, a, false, b, (short)0, c, false, false);
}

// stage W[f0+D]: all addresses = fixed base + compile-time immediates
template<int D>
__device__ __forceinline__ void stage_W(const float* __restrict__ wst,
                                        unsigned short* __restrict__ whb,
                                        unsigned short* __restrict__ wlb) {
#pragma unroll
    for (int i = 0; i < 4; ++i) {
        float v0 = wst[D * (KSZ * LFEAT) + i * 512];
        float v1 = wst[D * (KSZ * LFEAT) + i * 512 + LFEAT];
        unsigned hp, lp;
        split2(v0, v1, hp, lp);           // k-pair at fixed n
        *(unsigned*)(whb + 16 * i) = hp;
        *(unsigned*)(wlb + 16 * i) = lp;
    }
}

template<int D>
__device__ __forceinline__ void folds_loop(
    const float* __restrict__ wst,
    unsigned short* __restrict__ whb0, unsigned short* __restrict__ wlb0,
    unsigned short* __restrict__ whb1, unsigned short* __restrict__ wlb1,
    unsigned short (*sWh)[LFEAT * WSTR], unsigned short (*sWl)[LFEAT * WSTR],
    const v8us Xh[9], const v8us Xl[9],
    float* __restrict__ ob, const float* __restrict__ bb,
    int l15, int half)
{
    if constexpr (D < FPB) {
        if constexpr (D + 1 < FPB) {      // prefetch next fold's W into other buffer
            if constexpr (((D + 1) & 1) != 0) stage_W<D + 1>(wst, whb1, wlb1);
            else                              stage_W<D + 1>(wst, whb0, wlb0);
        }

        const unsigned short* swh = sWh[D & 1];
        const unsigned short* swl = sWl[D & 1];
        const unsigned short* bh0 = &swh[l15 * WSTR];
        const unsigned short* bl0 = &swl[l15 * WSTR];
        const unsigned short* bh1 = &swh[(l15 + 16) * WSTR];
        const unsigned short* bl1 = &swl[(l15 + 16) * WSTR];
        const int off = half * 8;

        v8f c0 = {}, c1 = {};
        {   // K = 0..31
            v16bf Ah = afrag<D + 0>(Xh);
            v16bf Al = afrag<D + 0>(Xl);
            v16bf Bh0 = ldB(bh0 + off),  Bl0 = ldB(bl0 + off);
            v16bf Bh1 = ldB(bh1 + off),  Bl1 = ldB(bl1 + off);
            c0 = bwmma(Ah, Bh0, c0);  c1 = bwmma(Ah, Bh1, c1);
            c0 = bwmma(Al, Bh0, c0);  c1 = bwmma(Al, Bh1, c1);
            c0 = bwmma(Ah, Bl0, c0);  c1 = bwmma(Ah, Bl1, c1);
        }
        {   // K = 32..63
            v16bf Ah = afrag<D + 32>(Xh);
            v16bf Al = afrag<D + 32>(Xl);
            v16bf Bh0 = ldB(bh0 + 32 + off),  Bl0 = ldB(bl0 + 32 + off);
            v16bf Bh1 = ldB(bh1 + 32 + off),  Bl1 = ldB(bl1 + 32 + off);
            c0 = bwmma(Ah, Bh0, c0);  c1 = bwmma(Ah, Bh1, c1);
            c0 = bwmma(Al, Bh0, c0);  c1 = bwmma(Al, Bh1, c1);
            c0 = bwmma(Ah, Bl0, c0);  c1 = bwmma(Ah, Bl1, c1);
        }

        // epilogue: base pointer + immediate offsets only
        float bias0 = bb[D * LFEAT];
        float bias1 = bb[D * LFEAT + 16];
        constexpr size_t RS = (size_t)FOLDS * LFEAT;
#pragma unroll
        for (int r = 0; r < 8; ++r) {
            ob[(size_t)D * LFEAT + (size_t)r * RS]      = c0[r] + bias0;
            ob[(size_t)D * LFEAT + (size_t)r * RS + 16] = c1[r] + bias1;
        }
        __syncthreads();
        folds_loop<D + 1>(wst, whb0, wlb0, whb1, wlb1, sWh, sWl, Xh, Xl,
                          ob, bb, l15, half);
    }
}

__global__ __launch_bounds__(256)
void local_linear_bf16x3(const float* __restrict__ x,
                         const float* __restrict__ w,
                         const float* __restrict__ bias,
                         float* __restrict__ out) {
    __shared__ unsigned short sXh[MTILE * XS];          // 22528 B
    __shared__ unsigned short sXl[MTILE * XS];          // 22528 B
    __shared__ unsigned short sWh[2][LFEAT * WSTR];     // 2 x 4608 B
    __shared__ unsigned short sWl[2][LFEAT * WSTR];     // 2 x 4608 B

    const int f0 = blockIdx.x * FPB;                    // multiple of 16
    const int b0 = blockIdx.y * MTILE;
    const int t  = threadIdx.x;

    // ---- stage x window: rows b0..b0+127, cols f0..f0+87 (split bf16) ----
    if (f0 + XS <= IN_F) {                 // fast path: aligned b128 loads
#pragma unroll
        for (int i = 0; i < 11; ++i) {
            int p = t + i * 256;           // 0..2815 = 128 rows x 22 quads
            int row = p / 22, qd = p - row * 22;
            int c0 = qd * 4;               // col, multiple of 4 -> 16B aligned
            v4f v = *(const v4f*)(x + (size_t)(b0 + row) * IN_F + f0 + c0);
            uv2 hp, lp;
            split4(v, hp, lp);
            *(uv2*)&sXh[row * XS + c0] = hp;
            *(uv2*)&sXl[row * XS + c0] = lp;
        }
    } else {                               // last fold column: guarded scalar path
#pragma unroll
        for (int i = 0; i < 22; ++i) {
            int p = t + i * 256;           // 0..5631 = 128 rows x 44 pairs
            int row = p / 44, cp = p - row * 44;
            int c0 = cp * 2;
            int g = f0 + c0;
            const float* xr = x + (size_t)(b0 + row) * IN_F;
            float v0 = (g     < IN_F) ? xr[g]     : 0.0f;
            float v1 = (g + 1 < IN_F) ? xr[g + 1] : 0.0f;
            unsigned hp, lp;
            split2(v0, v1, hp, lp);
            *(unsigned*)&sXh[row * XS + c0] = hp;
            *(unsigned*)&sXl[row * XS + c0] = lp;
        }
    }

    // ---- per-thread W staging bases (n invariant across iterations) ----
    const int n_st = t & 31, kp0 = t >> 5;              // kp0 in 0..7
    const float* wst = w + (size_t)f0 * (KSZ * LFEAT) + (2 * kp0) * LFEAT + n_st;
    unsigned short* whb0 = &sWh[0][n_st * WSTR + 2 * kp0];
    unsigned short* whb1 = &sWh[1][n_st * WSTR + 2 * kp0];
    unsigned short* wlb0 = &sWl[0][n_st * WSTR + 2 * kp0];
    unsigned short* wlb1 = &sWl[1][n_st * WSTR + 2 * kp0];

    stage_W<0>(wst, whb0, wlb0);
    __syncthreads();

    const int lane = t & 31;
    const int half = lane >> 4;
    const int l15  = lane & 15;
    const int m0   = (t >> 5) * 16;        // 8 waves x 16 rows = 128 rows

    // ---- lane-resident split x row, lane-local origin half*8 ----
    const unsigned short* xrh = &sXh[(m0 + l15) * XS + half * 8];
    const unsigned short* xrl = &sXl[(m0 + l15) * XS + half * 8];
    v8us Xh[9], Xl[9];
#pragma unroll
    for (int j = 0; j < 9; ++j) {
        Xh[j] = *(const v8us*)(xrh + 8 * j);
        Xl[j] = *(const v8us*)(xrl + 8 * j);
    }

    constexpr size_t RS = (size_t)FOLDS * LFEAT;
    float* ob = out + (size_t)(b0 + m0 + half * 8) * RS + (size_t)f0 * LFEAT + l15;
    const float* bb = bias + f0 * LFEAT + l15;

    folds_loop<0>(wst, whb0, wlb0, whb1, wlb1, sWh, sWl, Xh, Xl, ob, bb, l15, half);
}

extern "C" void kernel_launch(void* const* d_in, const int* in_sizes, int n_in,
                              void* d_out, int out_size, void* d_ws, size_t ws_size,
                              hipStream_t stream) {
    const float* x    = (const float*)d_in[0];  // (256, 4096)
    const float* w    = (const float*)d_in[1];  // (4096, 64, 32)
    const float* bias = (const float*)d_in[2];  // (4096, 32)
    float* out        = (float*)d_out;          // (256, 4096, 32)

    dim3 grid(FOLDS / FPB, 256 / MTILE);        // 256 x 2 blocks, 256 thr each
    local_linear_bf16x3<<<grid, 256, 0, stream>>>(x, w, bias, out);
}
